// GCNN_54254026883807
// MI455X (gfx1250) — compile-verified, matrix operand order checked by reading, not run on hardware
//
#include <hip/hip_runtime.h>
#include <hip/hip_bf16.h>

typedef __attribute__((ext_vector_type(16))) __bf16 v16bf;
typedef __attribute__((ext_vector_type(8)))  float  v8f;

#define H_DIM 128

__device__ __forceinline__ unsigned short f2bf(float x) {
  unsigned int u = __float_as_uint(x);
  unsigned int r = (u + 0x7FFFu + ((u >> 16) & 1u)) >> 16;  // RNE
  return (unsigned short)r;
}

union AFrag { v16bf v; unsigned short u[16]; uint4 q[2]; };

// A-fragment from f32 source (convert). Caller pre-offsets by half*8:
// elements 0..7 <- p[0..7] (k=kb+half*8..), 8..15 <- p[16..23] (k=kb+16+half*8..)
__device__ __forceinline__ v16bf make_a_frag_f32(const float* __restrict__ p) {
  AFrag a;
#pragma unroll
  for (int j = 0; j < 8; ++j) a.u[j] = f2bf(p[j]);
#pragma unroll
  for (int j = 0; j < 8; ++j) a.u[8 + j] = f2bf(p[16 + j]);
  return a.v;
}

// A-fragment from bf16 source: two aligned 16B loads, no ALU.
__device__ __forceinline__ v16bf make_a_frag_bf16(const unsigned short* __restrict__ p) {
  AFrag a;
  a.q[0] = *(const uint4*)(p);        // k = kb+half*8 .. +7
  a.q[1] = *(const uint4*)(p + 16);   // k = kb+16+half*8 .. +7
  return a.v;
}

// Cooperative block copy of packed B-fragments into LDS (16B per lane-iter).
// blockDim.x must be 256. Includes the barrier.
__device__ __forceinline__ void stage_frags(const unsigned short* __restrict__ g,
                                            unsigned short* l, int total_u16) {
  const int iters = total_u16 >> 3;  // 8 u16 = 16 bytes per iteration
  for (int i = threadIdx.x; i < iters; i += 256)
    *(uint4*)(l + (size_t)i * 8) = *(const uint4*)(g + (size_t)i * 8);
  __syncthreads();
}

// ---------------------------------------------------------------------------
// Pack a KxNcols f32 weight matrix into per-lane bf16 B-fragments.
// Fragment (ks,nt): 32 lanes x 16 bf16. lane holds column nt*16+(lane&15),
// k = ks*32 + ((j&8)?16:0) + (lane>>4)*8 + (j&7).
// ---------------------------------------------------------------------------
__global__ void pack_bfrag(const float* __restrict__ W, int ncols, int NT,
                           unsigned short* __restrict__ out, int total) {
  int tid = blockIdx.x * blockDim.x + threadIdx.x;
  if (tid >= total) return;
  int j    = tid & 15;
  int lane = (tid >> 4) & 31;
  int frag = tid >> 9;
  int nt   = frag % NT;
  int ks   = frag / NT;
  int col  = nt * 16 + (lane & 15);
  int half = lane >> 4;
  int k    = ks * 32 + ((j & 8) ? 16 : 0) + half * 8 + (j & 7);
  out[tid] = f2bf(W[(size_t)k * ncols + col]);
}

// ---------------------------------------------------------------------------
// Generic row-tiled WMMA GEMM: Out[nrows x (NT*16)] = act(A[nrows x K] @ B + b)
// 256 threads = 8 waves; each wave computes a 16-row stripe. B fragments are
// staged once per block into LDS and shared by all 8 waves.
// A_BF16: A is row-major bf16 (no convert); else f32. OUT_BF16: store bf16.
// ---------------------------------------------------------------------------
template <int KSTEPS, int NT, bool RELU, bool HASBIAS, bool A_BF16, bool OUT_BF16>
__global__ __launch_bounds__(256)
void gemm_wmma_bf16(const void* __restrict__ Av,
                    const unsigned short* __restrict__ Bfrag,
                    const float* __restrict__ bias,
                    void* __restrict__ Outv, int nrows) {
  constexpr int K    = KSTEPS * 32;
  constexpr int NCOL = NT * 16;
  __shared__ unsigned short bs[KSTEPS * NT * 512];   // <= 32 KB

  const int lane = threadIdx.x & 31;
  const int wave = threadIdx.x >> 5;
  const int rowBase = blockIdx.x * 128 + wave * 16;
  const int m = lane & 15;
  const int half = lane >> 4;
  int arow = rowBase + m;
  if (arow >= nrows) arow = nrows - 1;           // clamp: keep EXEC all-ones
  const float*          apf = (const float*)Av          + (size_t)arow * K + half * 8;
  const unsigned short* apb = (const unsigned short*)Av + (size_t)arow * K + half * 8;
  if (A_BF16) __builtin_prefetch(apb, 0, 1);     // global_prefetch_b8
  else        __builtin_prefetch(apf, 0, 1);

  stage_frags(Bfrag, bs, KSTEPS * NT * 512);

  v8f acc[NT];
#pragma unroll
  for (int nt = 0; nt < NT; ++nt) acc[nt] = (v8f)(0.0f);

#pragma unroll
  for (int ks = 0; ks < KSTEPS; ++ks) {
    v16bf a = A_BF16 ? make_a_frag_bf16(apb + ks * 32)
                     : make_a_frag_f32(apf + ks * 32);
#pragma unroll
    for (int nt = 0; nt < NT; ++nt) {
      v16bf b = *(const v16bf*)(bs + ((ks * NT + nt) * 32 + lane) * 16);
      acc[nt] = __builtin_amdgcn_wmma_f32_16x16x32_bf16(
          false, a, false, b, (short)0, acc[nt], false, false);
    }
  }

#pragma unroll
  for (int nt = 0; nt < NT; ++nt) {
    int col = nt * 16 + m;
    float bv = HASBIAS ? bias[col] : 0.0f;
#pragma unroll
    for (int r = 0; r < 8; ++r) {
      int orow = rowBase + r + half * 8;
      if (orow < nrows) {
        float v = acc[nt][r] + bv;
        if (RELU) v = fmaxf(v, 0.0f);
        size_t idx = (size_t)orow * NCOL + col;
        if (OUT_BF16) ((unsigned short*)Outv)[idx] = f2bf(v);
        else          ((float*)Outv)[idx] = v;
      }
    }
  }
}

// ------------------------------ GCN scatter --------------------------------
__global__ void deg_init(float* __restrict__ deg, int n) {
  int i = blockIdx.x * blockDim.x + threadIdx.x;
  if (i < n) deg[i] = 1.0f;                      // self loop
}
__global__ void deg_count(const int* __restrict__ tgt, float* __restrict__ deg, int e) {
  int i = blockIdx.x * blockDim.x + threadIdx.x;
  if (i < e) atomicAdd(&deg[tgt[i]], 1.0f);
}
__global__ void deg_rsqrt(float* __restrict__ deg, int n) {
  int i = blockIdx.x * blockDim.x + threadIdx.x;
  if (i < n) deg[i] = rsqrtf(deg[i]);
}
// agg[i,:] = xw[i,:] * dinv[i]^2   (self-loop contribution, also init)
__global__ void agg_self(const float* __restrict__ xw, const float* __restrict__ dinv,
                         float* __restrict__ agg, int n) {
  long long t = (long long)blockIdx.x * blockDim.x + threadIdx.x;
  if (t >= (long long)n * H_DIM) return;
  int i = (int)(t >> 7);
  float d = dinv[i];
  agg[t] = xw[t] * d * d;
}
// agg[t,:] += xw[s,:] * dinv[s]*dinv[t]   (32 lanes/edge, float4 each)
__global__ void scatter_edges(const int* __restrict__ src, const int* __restrict__ tgt,
                              const float* __restrict__ xw, const float* __restrict__ dinv,
                              float* __restrict__ agg, int e) {
  long long t = (long long)blockIdx.x * blockDim.x + threadIdx.x;
  if (t >= (long long)e * 32) return;
  int ei = (int)(t >> 5);
  int l  = (int)(t & 31);
  int s  = src[ei], d = tgt[ei];
  float c = dinv[s] * dinv[d];
  const float4 v = *(const float4*)(xw + (size_t)s * H_DIM + l * 4);
  float* o = agg + (size_t)d * H_DIM + l * 4;
  atomicAdd(o + 0, v.x * c);
  atomicAdd(o + 1, v.y * c);
  atomicAdd(o + 2, v.z * c);
  atomicAdd(o + 3, v.w * c);
}
// h = relu(agg + b), stored as bf16 (feeds only WMMA A-sides)
__global__ void bias_relu_bf16(const float* __restrict__ agg, const float* __restrict__ b,
                               unsigned short* __restrict__ out, int n) {
  long long t = (long long)blockIdx.x * blockDim.x + threadIdx.x;
  if (t >= (long long)n * H_DIM) return;
  out[t] = f2bf(fmaxf(agg[t] + b[t & (H_DIM - 1)], 0.0f));
}

// ------------------------- node head: 128->64->12 --------------------------
__global__ __launch_bounds__(256)
void node_head(const unsigned short* __restrict__ h, const unsigned short* __restrict__ Bfrag,
               const float* __restrict__ b1, const float* __restrict__ W2,  // 64x12
               const float* __restrict__ b2, float* __restrict__ out, int nrows) {
  __shared__ unsigned short bs[16 * 512];        // 16 KB of Wn1 fragments
  __shared__ float act[8][16][64];               // 32 KB
  const int lane = threadIdx.x & 31;
  const int wave = threadIdx.x >> 5;
  const int rowBase = blockIdx.x * 128 + wave * 16;
  const int m = lane & 15, half = lane >> 4;
  int arow = rowBase + m;
  if (arow >= nrows) arow = nrows - 1;
  const unsigned short* ap = h + (size_t)arow * H_DIM + half * 8;
  __builtin_prefetch(ap, 0, 1);

  stage_frags(Bfrag, bs, 16 * 512);

  v8f acc[4];
#pragma unroll
  for (int nt = 0; nt < 4; ++nt) acc[nt] = (v8f)(0.0f);
#pragma unroll
  for (int ks = 0; ks < 4; ++ks) {
    v16bf a = make_a_frag_bf16(ap + ks * 32);
#pragma unroll
    for (int nt = 0; nt < 4; ++nt) {
      v16bf b = *(const v16bf*)(bs + ((ks * 4 + nt) * 32 + lane) * 16);
      acc[nt] = __builtin_amdgcn_wmma_f32_16x16x32_bf16(
          false, a, false, b, (short)0, acc[nt], false, false);
    }
  }
#pragma unroll
  for (int nt = 0; nt < 4; ++nt) {
    int col = nt * 16 + m;
    float bb = b1[col];
#pragma unroll
    for (int r = 0; r < 8; ++r)
      act[wave][r + half * 8][col] = fmaxf(acc[nt][r] + bb, 0.0f);
  }
  __syncthreads();
#pragma unroll
  for (int i = 0; i < 6; ++i) {                  // 32 lanes * 6 = 16 nodes * 12
    int idx = lane * 6 + i;
    int nl = idx / 12, c = idx - nl * 12;
    int node = rowBase + nl;
    float s = b2[c];
    for (int k = 0; k < 64; ++k) s += act[wave][nl][k] * W2[k * 12 + c];
    if (node < nrows) out[(size_t)node * 12 + c] = s;
  }
}

// --------------- edge head: [h[row]|h[col]|attr](288) ->128-> 4 ------------
// h gathered as bf16 (no convert ALU, half the bytes). B fragments for
// ksteps 0..7 staged in LDS (64 KB); kstep 8 (attr, f32) read from global.
// Second stage (128->4) via in-register partials + xor-shuffle reduction.
__global__ __launch_bounds__(256)
void edge_head(const unsigned short* __restrict__ h, const int* __restrict__ ei,
               const float* __restrict__ eattr, const unsigned short* __restrict__ Bfrag,
               const float* __restrict__ b1, const float* __restrict__ W2,  // 128x4
               const float* __restrict__ b2, float* __restrict__ out, int E_) {
  __shared__ unsigned short bs[8 * 8 * 512];     // 64 KB: ksteps 0..7
  const int lane = threadIdx.x & 31;
  const int wave = threadIdx.x >> 5;
  const int eBase = blockIdx.x * 128 + wave * 16;
  const int m = lane & 15, half = lane >> 4;
  int e = eBase + m;
  if (e >= E_) e = E_ - 1;
  const int rs = ei[e];
  const int rt = ei[E_ + e];
  const unsigned short* ph = h + (size_t)rs * H_DIM;
  const unsigned short* pt = h + (size_t)rt * H_DIM;
  const float* pa = eattr + (size_t)e * 32;
  __builtin_prefetch(ph, 0, 1);                  // gathered rows: prefetch
  __builtin_prefetch(pt, 0, 1);
  __builtin_prefetch(pa, 0, 1);

  stage_frags(Bfrag, bs, 8 * 8 * 512);

  v8f acc[8];
#pragma unroll
  for (int nt = 0; nt < 8; ++nt) acc[nt] = (v8f)(0.0f);

#pragma unroll
  for (int ks = 0; ks < 9; ++ks) {               // K = 288
    const int kb = ks * 32;
    v16bf a;
    if (ks < 8) {
      const unsigned short* p = (kb < 128) ? (ph + kb) : (pt + (kb - 128));
      a = make_a_frag_bf16(p + half * 8);
    } else {
      a = make_a_frag_f32(pa + half * 8);
    }
#pragma unroll
    for (int nt = 0; nt < 8; ++nt) {
      v16bf b = (ks < 8)
          ? *(const v16bf*)(bs + ((ks * 8 + nt) * 32 + lane) * 16)
          : *(const v16bf*)(Bfrag + ((size_t)(8 * 8 + nt) * 32 + lane) * 16);
      acc[nt] = __builtin_amdgcn_wmma_f32_16x16x32_bf16(
          false, a, false, b, (short)0, acc[nt], false, false);
    }
  }

  // fused 128->4: per-lane partials over this lane's hidden column, then
  // xor-shuffle reduction across the 16 lanes of each half.
  float part[8][4];
#pragma unroll
  for (int r = 0; r < 8; ++r)
#pragma unroll
    for (int c = 0; c < 4; ++c) part[r][c] = 0.0f;

#pragma unroll
  for (int nt = 0; nt < 8; ++nt) {
    int col = nt * 16 + m;
    float bb = b1[col];
    float4 w = *(const float4*)(W2 + col * 4);
#pragma unroll
    for (int r = 0; r < 8; ++r) {
      float hv = fmaxf(acc[nt][r] + bb, 0.0f);
      part[r][0] += hv * w.x;
      part[r][1] += hv * w.y;
      part[r][2] += hv * w.z;
      part[r][3] += hv * w.w;
    }
  }
#pragma unroll
  for (int off = 1; off < 16; off <<= 1) {
#pragma unroll
    for (int r = 0; r < 8; ++r)
#pragma unroll
      for (int c = 0; c < 4; ++c)
        part[r][c] += __shfl_xor(part[r][c], off, 32);
  }
  // each half-lane stores 2 of its half's 32 outputs (8 rows x 4 classes)
#pragma unroll
  for (int i = 0; i < 2; ++i) {
    int idx = m * 2 + i;
    int r = idx >> 2, c = idx & 3;
    float sel = 0.0f;
#pragma unroll
    for (int rr = 0; rr < 8; ++rr)
#pragma unroll
      for (int cc = 0; cc < 4; ++cc)
        if (rr == r && cc == c) sel = part[rr][cc];
    int eo = eBase + half * 8 + r;
    if (eo < E_) out[(size_t)eo * 4 + c] = sel + b2[c];
  }
}

// ---------------------------------------------------------------------------
extern "C" void kernel_launch(void* const* d_in, const int* in_sizes, int n_in,
                              void* d_out, int out_size, void* d_ws, size_t ws_size,
                              hipStream_t stream) {
  const float* x      = (const float*)d_in[0];
  const int*   ei     = (const int*)d_in[1];
  const float* eattr  = (const float*)d_in[2];
  const float* We     = (const float*)d_in[3];
  const float* be     = (const float*)d_in[4];
  const float* Wc1    = (const float*)d_in[5];
  const float* bc1    = (const float*)d_in[6];
  const float* Wc2    = (const float*)d_in[7];
  const float* bc2    = (const float*)d_in[8];
  const float* Wn1    = (const float*)d_in[9];
  const float* bn1    = (const float*)d_in[10];
  const float* Wn2    = (const float*)d_in[11];
  const float* bn2    = (const float*)d_in[12];
  const float* Wedge1 = (const float*)d_in[13];
  const float* bedge1 = (const float*)d_in[14];
  const float* Wedge2 = (const float*)d_in[15];
  const float* bedge2 = (const float*)d_in[16];

  const int N  = in_sizes[0] / 64;
  const int E_ = in_sizes[1] / 2;

  // workspace: xw(f32 NxH), agg(f32 NxH), deg(f32 N), hb(bf16 NxH), frags
  char* base = (char*)d_ws;
  float* xw  = (float*)base;
  float* agg = xw  + (size_t)N * H_DIM;
  float* deg = agg + (size_t)N * H_DIM;
  size_t off1 = (((size_t)(2 * (size_t)N * H_DIM + N) * 4 + 255) & ~(size_t)255);
  unsigned short* hb = (unsigned short*)(base + off1);
  size_t off2 = ((off1 + (size_t)N * H_DIM * 2 + 255) & ~(size_t)255);
  unsigned short* fWe  = (unsigned short*)(base + off2);
  unsigned short* fWc1 = fWe  + 2 * 8 * 512;
  unsigned short* fWc2 = fWc1 + 4 * 8 * 512;
  unsigned short* fWn1 = fWc2 + 4 * 8 * 512;
  unsigned short* fWeg = fWn1 + 4 * 4 * 512;

  int t;
  t = 2 * 8 * 512; pack_bfrag<<<(t + 255) / 256, 256, 0, stream>>>(We, 128, 8, fWe, t);
  t = 4 * 8 * 512; pack_bfrag<<<(t + 255) / 256, 256, 0, stream>>>(Wc1, 128, 8, fWc1, t);
  t = 4 * 8 * 512; pack_bfrag<<<(t + 255) / 256, 256, 0, stream>>>(Wc2, 128, 8, fWc2, t);
  t = 4 * 4 * 512; pack_bfrag<<<(t + 255) / 256, 256, 0, stream>>>(Wn1, 64, 4, fWn1, t);
  t = 9 * 8 * 512; pack_bfrag<<<(t + 255) / 256, 256, 0, stream>>>(Wedge1, 128, 8, fWeg, t);

  // degrees (shared by both convs)
  deg_init <<<(N  + 255) / 256, 256, 0, stream>>>(deg, N);
  deg_count<<<(E_ + 255) / 256, 256, 0, stream>>>(ei + E_, deg, E_);
  deg_rsqrt<<<(N  + 255) / 256, 256, 0, stream>>>(deg, N);

  const int gGrid = (N + 127) / 128;
  const long long nh = (long long)N * H_DIM;
  const int nhGrid = (int)((nh + 255) / 256);
  const int scGrid = (int)(((long long)E_ * 32 + 255) / 256);

  // h0 = relu(relu(x @ We + be))  (relu idempotent), stored bf16
  gemm_wmma_bf16<2, 8, true, true, false, true>
      <<<gGrid, 256, 0, stream>>>(x, fWe, be, hb, N);

  // conv1: xw = h0 @ Wc1 ; agg = self + scatter ; h1 = relu(agg + bc1)
  gemm_wmma_bf16<4, 8, false, false, true, false>
      <<<gGrid, 256, 0, stream>>>(hb, fWc1, nullptr, xw, N);
  agg_self      <<<nhGrid, 256, 0, stream>>>(xw, deg, agg, N);
  scatter_edges <<<scGrid, 256, 0, stream>>>(ei, ei + E_, xw, deg, agg, E_);
  bias_relu_bf16<<<nhGrid, 256, 0, stream>>>(agg, bc1, hb, N);

  // conv2
  gemm_wmma_bf16<4, 8, false, false, true, false>
      <<<gGrid, 256, 0, stream>>>(hb, fWc2, nullptr, xw, N);
  agg_self      <<<nhGrid, 256, 0, stream>>>(xw, deg, agg, N);
  scatter_edges <<<scGrid, 256, 0, stream>>>(ei, ei + E_, xw, deg, agg, E_);
  bias_relu_bf16<<<nhGrid, 256, 0, stream>>>(agg, bc2, hb, N);

  float* nodeOut = (float*)d_out;
  float* edgeOut = nodeOut + (size_t)N * 12;

  node_head<<<gGrid, 256, 0, stream>>>(hb, fWn1, bn1, Wn2, bn2, nodeOut, N);
  edge_head<<<(E_ + 127) / 128, 256, 0, stream>>>(hb, ei, eattr, fWeg, bedge1,
                                                  Wedge2, bedge2, edgeOut, E_);
}
